// CausalKernel_71631464563102
// MI455X (gfx1250) — matrix operand order repaired and checked by reading, not compiled, not run on hardware
//
#include <hip/hip_runtime.h>

// ---------------------------------------------------------------------------
// CausalKernel for MI455X (gfx1250, wave32, WMMA + async LDS staging)
//
// spatial mode sum  S(r) = sum_{n<35937} k[n] sin(n r)
// reformulated with n = 32*g + h:
//   S(r) = sum_g [ sin(32 g r) * P(g) + cos(32 g r) * Q(g) ]
//   P(g) = sum_h k[32g+h] cos(h r)   (GEMM vs shared K matrix)
//   Q(g) = sum_h k[32g+h] sin(h r)
// P/Q via v_wmma_f32_16x16x32_f16 (M=16 points, K=32, N=16 groups); outer
// phases via per-point rotation recurrence (step angle 512*r, v_pk_fma_f32).
// Shared coefficient matrix staged to LDS with GLOBAL_LOAD_ASYNC_TO_LDS_B128
// (ASYNCcnt path) instead of VGPR round trips.
// ---------------------------------------------------------------------------

typedef __attribute__((ext_vector_type(16))) _Float16 v16h;
typedef __attribute__((ext_vector_type(8)))  float    v8f;

#define N_PTS      131072
#define M_SPATIAL  35937
#define NBLK       71            // 71 blocks * 16 groups * 32 modes = 36352 (zero padded)
#define PACKED_H   (NBLK * 512)  // halves in packed B
#define CHUNK_BLK  32            // B blocks staged per LDS round (32 KB)

// --- pack spatial kernel into f16 WMMA B-matrix layout ----------------------
// B is 32(K) x 16(N) per block. Contiguous k order used here: lanes 0-15 hold K=0..15 (VGPR j ->
// K=2j,2j+1), lanes 16-31 hold K=16..31; N = lane%16. Stored so a lane can
// load its v16h as 32 contiguous bytes: packed[b*512 + lane*16 + t].
__global__ void pack_kernel(const float* __restrict__ k, _Float16* __restrict__ packed) {
    int idx = blockIdx.x * blockDim.x + threadIdx.x;
    if (idx >= PACKED_H) return;
    int b = idx >> 9;            // block
    int L = (idx >> 4) & 31;     // lane
    int t = idx & 15;            // half within lane
    int n = L & 15;              // column (group within block)
    int kk = ((L >> 4) << 4) + t;// K row 0..31
    int src = 32 * (16 * b + n) + kk;
    float v = (src < M_SPATIAL) ? k[src] : 0.0f;
    packed[idx] = (_Float16)v;
}

// --- WMMA spatial mode sum --------------------------------------------------
__global__ void __launch_bounds__(256) spatial_wmma_kernel(
        const float* __restrict__ coords,
        const _Float16* __restrict__ packedB,
        float* __restrict__ spatial_sum) {
    __shared__ __align__(64) _Float16 ldsB[CHUNK_BLK * 512]; // 32 KB

    const int tid  = threadIdx.x;
    const int lane = tid & 31;
    const int wave = tid >> 5;
    const int tile = blockIdx.x * 8 + wave;   // 16 points per wave
    const int base = tile * 16;

    // r of this lane's A-matrix row (point m = lane%16)
    const int m = lane & 15;
    float4 cm = ((const float4*)coords)[base + m];
    float r_m = sqrtf(cm.y * cm.y + cm.z * cm.z + cm.w * cm.w + 1e-12f);

    // A matrices: cos(h r_m), sin(h r_m) in contiguous-k A 16x32 layout (same k order as packed B).
    // lanes 0-15: t=0..15 -> h=0..15
    // lanes 16-31: t=0..15 -> h=16..31
    v16h Acos, Asin;
    const int hbase = (lane < 16) ? 0 : 16;
    #pragma unroll
    for (int t = 0; t < 16; ++t) {
        int h = hbase + t;
        float sh, ch;
        __sincosf((float)h * r_m, &sh, &ch);
        Acos[t] = (_Float16)ch;
        Asin[t] = (_Float16)sh;
    }

    // Epilogue rotation state: D tile rows are points rowbase..rowbase+7,
    // column = lane%16 -> group g = blk*16 + (lane%16); per block g += 16,
    // i.e. phase step 512*r per row.
    const int rowbase = (lane < 16) ? 0 : 8;
    float s[8], c[8], stS[8], stC[8], acc[8];
    #pragma unroll
    for (int j = 0; j < 8; ++j) {
        float rj = __shfl(r_m, rowbase + j, 32);
        __sincosf(32.0f * (float)(lane & 15) * rj, &s[j], &c[j]);
        __sincosf(512.0f * rj, &stS[j], &stC[j]);
        acc[j] = 0.0f;
    }

    const unsigned lds_base = (unsigned)(unsigned long long)(&ldsB[0]);

    for (int cs = 0; cs < NBLK; cs += CHUNK_BLK) {
        int nb = NBLK - cs; if (nb > CHUNK_BLK) nb = CHUNK_BLK;
        __syncthreads();   // previous chunk fully consumed before overwrite
        // async stage of shared coefficients: memory -> LDS, no VGPR round trip
        {
            const char* gbase = (const char*)(packedB + cs * 512);
            const int bytes = nb * 1024;
            for (int e = tid * 16; e < bytes; e += 256 * 16) {
                unsigned lds_off = lds_base + (unsigned)e;
                unsigned long long gaddr = (unsigned long long)(gbase + e);
                asm volatile("global_load_async_to_lds_b128 %0, %1, off"
                             :: "v"(lds_off), "v"(gaddr)
                             : "memory");
            }
#if __has_builtin(__builtin_amdgcn_s_wait_asynccnt)
            __builtin_amdgcn_s_wait_asynccnt(0);
#else
            asm volatile("s_wait_asynccnt 0x0" ::: "memory");
#endif
        }
        __syncthreads();

        for (int b = 0; b < nb; ++b) {
            v16h Bv = *(const v16h*)(ldsB + b * 512 + lane * 16);
            v8f zero = {};
            // P = Acos x K, Q = Asin x K  (f32 accumulate, C = 0)
            v8f P = __builtin_amdgcn_wmma_f32_16x16x32_f16(
                        false, Acos, false, Bv, (short)0, zero, false, false);
            v8f Q = __builtin_amdgcn_wmma_f32_16x16x32_f16(
                        false, Asin, false, Bv, (short)0, zero, false, false);
            #pragma unroll
            for (int j = 0; j < 8; ++j) {
                acc[j] = fmaf(s[j], P[j], fmaf(c[j], Q[j], acc[j]));
                float ns = fmaf(s[j], stC[j],  c[j] * stS[j]);
                float nc = fmaf(c[j], stC[j], -s[j] * stS[j]);
                s[j] = ns; c[j] = nc;
            }
        }
    }

    // reduce over the 16 columns (lanes 0-15 / 16-31 hold disjoint rows)
    #pragma unroll
    for (int j = 0; j < 8; ++j) {
        acc[j] += __shfl_xor(acc[j], 1, 32);
        acc[j] += __shfl_xor(acc[j], 2, 32);
        acc[j] += __shfl_xor(acc[j], 4, 32);
        acc[j] += __shfl_xor(acc[j], 8, 32);
    }
    if ((lane & 15) == 0) {
        #pragma unroll
        for (int j = 0; j < 8; ++j)
            spatial_sum[base + rowbase + j] = acc[j];
    }
}

// --- elementwise finalize ---------------------------------------------------
__global__ void finalize_kernel(const float* __restrict__ coords,
                                const float* __restrict__ temporal_kernel,
                                const float* __restrict__ massp,
                                const float* __restrict__ coupp,
                                const float* __restrict__ spatial_sum,
                                float* __restrict__ out) {
    int i = blockIdx.x * blockDim.x + threadIdx.x;
    if (i >= N_PTS) return;
    float4 cc = ((const float4*)coords)[i];
    float t = cc.x, x = cc.y, y = cc.z, z = cc.w;
    float d2 = x * x + y * y + z * z;
    float t2 = t * t;
    float interval = t2 - d2;
    bool timelike  = interval > 0.0f;
    bool lightlike = fabsf(interval) < 1e-10f;

    float dist = sqrtf(d2 + 1e-12f);
    float mass = massp[0];
    float coup = coupp[0];
    float green = (timelike && t > 0.0f) ? coup * __expf(-mass * dist) / dist : 0.0f;
    float kv = (timelike || lightlike) ? green : 0.0f;

    float spatial = spatial_sum[i] / (dist + 1e-6f);

    float temporal = 0.0f;
    #pragma unroll
    for (int nn = 0; nn < 33; ++nn)
        temporal = fmaf(__cosf(t * (float)(nn + 1) * 0.1f), temporal_kernel[nn], temporal);
    temporal *= __expf(-0.1f * fabsf(t));

    float total = kv + spatial * temporal;
    bool future = (t2 >= d2) && (t >= 0.0f);
    out[i] = future ? total : 0.0f;
}

// --- launcher ---------------------------------------------------------------
extern "C" void kernel_launch(void* const* d_in, const int* in_sizes, int n_in,
                              void* d_out, int out_size, void* d_ws, size_t ws_size,
                              hipStream_t stream) {
    const float* coords = (const float*)d_in[0];
    const float* skern  = (const float*)d_in[1];
    const float* tkern  = (const float*)d_in[2];
    const float* mass   = (const float*)d_in[3];
    const float* coup   = (const float*)d_in[4];
    float* out = (float*)d_out;

    float*     spatial_sum = (float*)d_ws;                                  // N floats
    _Float16*  packedB     = (_Float16*)((char*)d_ws + N_PTS * sizeof(float));

    pack_kernel<<<(PACKED_H + 255) / 256, 256, 0, stream>>>(skern, packedB);
    spatial_wmma_kernel<<<N_PTS / (16 * 8), 256, 0, stream>>>(coords, packedB, spatial_sum);
    finalize_kernel<<<(N_PTS + 255) / 256, 256, 0, stream>>>(coords, tkern, mass, coup,
                                                             spatial_sum, out);
}
